// HeteroGNN_6030134084334
// MI455X (gfx1250) — matrix operand order repaired and checked by reading, not compiled, not run on hardware
//
#include <hip/hip_runtime.h>
#include <hip/hip_bf16.h>

// ---------------------------------------------------------------------------
// HeteroGNN forward for MI455X (gfx1250, wave32, WMMA).
// All dense 64/128-K GEMMs use v_wmma_f32_16x16x32_bf16 (weights pre-converted
// to bf16, transposed to Wt[col][k]); everything else is bandwidth-bound
// edge scatter/softmax and elementwise BN/ReLU passes.
// ---------------------------------------------------------------------------

#define H64 64

typedef __bf16 v16bf __attribute__((ext_vector_type(16)));
typedef float  v8f   __attribute__((ext_vector_type(8)));

static __device__ __forceinline__ __bf16 f2bf(float f) {
  unsigned u = __float_as_uint(f);
  unsigned short h = (unsigned short)(u >> 16);
  __bf16 r;
  __builtin_memcpy(&r, &h, 2);
  return r;
}

// ordered-uint encoding for float atomicMax
static __device__ __forceinline__ unsigned fenc(float f) {
  unsigned u = __float_as_uint(f);
  return (u & 0x80000000u) ? ~u : (u | 0x80000000u);
}
static __device__ __forceinline__ float fdec(unsigned u) {
  unsigned v = (u & 0x80000000u) ? (u & 0x7FFFFFFFu) : ~u;
  return __uint_as_float(v);
}

// --------------------------- WMMA GEMM -------------------------------------
// C[nrows,64] = [A0 | A1][nrows, 32*ktiles] @ W   (W given as bf16 Wt[col][k])
// One wave -> 16 rows x 64 cols, 4*ktiles WMMAs.
__global__ void __launch_bounds__(256)
k_gemm_wmma(const float* __restrict__ A0, const float* __restrict__ A1,
            const __bf16* __restrict__ Wt, float* __restrict__ C,
            int nrows, int ktiles) {
  const int wave = threadIdx.x >> 5;
  const int lane = threadIdx.x & 31;
  const int r0 = (blockIdx.x * 8 + wave) * 16;
  if (r0 >= nrows) return;                 // uniform per wave: EXEC stays all-1
  const int m  = lane & 15;
  const int hi = lane >> 4;
  const int Ktot = ktiles * 32;
  const size_t cstep = (size_t)16 * Ktot;

  v8f c0 = {}, c1 = {}, c2 = {}, c3 = {};
  const int row = r0 + m;

  for (int kt = 0; kt < ktiles; ++kt) {
    const float* Asrc = (kt < 2) ? A0 : A1;
    const int kb = (kt & 1) * 32;
    const float* ap = Asrc + (size_t)row * H64 + kb + 16 * hi;
    float af[16];
#pragma unroll
    for (int j = 0; j < 8; ++j) af[j] = ap[j];        // K = kb+16hi+j
#pragma unroll
    for (int j = 0; j < 8; ++j) af[8 + j] = ap[8 + j]; // K = kb+16hi+8+j
    v16bf a;
#pragma unroll
    for (int j = 0; j < 16; ++j) a[j] = f2bf(af[j]);

    const __bf16* wb = Wt + (size_t)m * Ktot + kt * 32 + 16 * hi;
    v16bf b0 = *(const v16bf*)(wb);
    v16bf b1 = *(const v16bf*)(wb + cstep);
    v16bf b2 = *(const v16bf*)(wb + 2 * cstep);
    v16bf b3 = *(const v16bf*)(wb + 3 * cstep);

    c0 = __builtin_amdgcn_wmma_f32_16x16x32_bf16(false, a, false, b0, (short)0, c0, false, false);
    c1 = __builtin_amdgcn_wmma_f32_16x16x32_bf16(false, a, false, b1, (short)0, c1, false, false);
    c2 = __builtin_amdgcn_wmma_f32_16x16x32_bf16(false, a, false, b2, (short)0, c2, false, false);
    c3 = __builtin_amdgcn_wmma_f32_16x16x32_bf16(false, a, false, b3, (short)0, c3, false, false);
  }

#pragma unroll
  for (int v = 0; v < 8; ++v) {
    float* crow = C + (size_t)(r0 + v + 8 * hi) * H64 + m;
    crow[0]  = c0[v];
    crow[16] = c1[v];
    crow[32] = c2[v];
    crow[48] = c3[v];
  }
}

// ------------------------- small helper kernels ----------------------------

// Wt[col*K + k] = (bf16) W[k*64 + col]
__global__ void k_transpose_bf16(const float* __restrict__ W, __bf16* __restrict__ Wt, int K) {
  int i = blockIdx.x * blockDim.x + threadIdx.x;
  if (i >= 64 * K) return;
  int col = i / K, k = i - col * K;
  Wt[(size_t)col * K + k] = f2bf(W[(size_t)k * 64 + col]);
}

// scalar sum / sumsq of a 1-D array into st[0], st[1]
__global__ void k_scalar_stats(const float* __restrict__ x, int n, float* __restrict__ st) {
  float ls = 0.f, lq = 0.f;
  for (size_t i = blockIdx.x * (size_t)blockDim.x + threadIdx.x; i < (size_t)n;
       i += (size_t)gridDim.x * blockDim.x) {
    float v = x[i]; ls += v; lq += v * v;
  }
  __shared__ float s[2];
  if (threadIdx.x == 0) { s[0] = 0.f; s[1] = 0.f; }
  __syncthreads();
  atomicAdd(&s[0], ls); atomicAdd(&s[1], lq);
  __syncthreads();
  if (threadIdx.x == 0) { atomicAdd(&st[0], s[0]); atomicAdd(&st[1], s[1]); }
}

// encoder: X[n,64] = relu(BN(x*W)) using rank-1 BN stats (bias cancels in BN)
__global__ void k_encode(const float* __restrict__ x, const float* __restrict__ W,
                         const float* __restrict__ g, const float* __restrict__ beta,
                         const float* __restrict__ st, int n, float* __restrict__ X) {
  size_t i = blockIdx.x * (size_t)blockDim.x + threadIdx.x;
  if (i >= (size_t)n * H64) return;
  int col = (int)(i & 63);
  float inv = 1.0f / (float)n;
  float mu = st[0] * inv;
  float var = st[1] * inv - mu * mu;
  float w = W[col];
  float c = (x[i >> 6] - mu) * w;
  float v = g[col] * c * rsqrtf(var * w * w + 1e-5f) + beta[col];
  X[i] = v > 0.f ? v : 0.f;
}

// wad[i] = sum_j W[i][j] * a[j]   (for s_d = x_d . (W a_d))
__global__ void k_matvec64(const float* __restrict__ W, const float* __restrict__ a,
                           float* __restrict__ wad) {
  int i = threadIdx.x;
  float acc = 0.f;
#pragma unroll 8
  for (int j = 0; j < 64; ++j) acc += W[(size_t)i * 64 + j] * a[j];
  wad[i] = acc;
}

// s[r] = X[r,:] . v
__global__ void k_rowdot(const float* __restrict__ X, const float* __restrict__ v,
                         float* __restrict__ s, int n) {
  int r = blockIdx.x * blockDim.x + threadIdx.x;
  if (r >= n) return;
  const float4* a = (const float4*)(X + (size_t)r * H64);
  const float4* vv = (const float4*)v;
  float acc = 0.f;
#pragma unroll
  for (int j = 0; j < 16; ++j) {
    float4 xa = a[j], xb = vv[j];
    acc += xa.x * xb.x + xa.y * xb.y + xa.z * xb.z + xa.w * xb.w;
  }
  s[r] = acc;
}

__global__ void k_edge_max(const int* __restrict__ src, const int* __restrict__ dst,
                           const float* __restrict__ ss, const float* __restrict__ sd,
                           unsigned* __restrict__ mxu, int E) {
  int e = blockIdx.x * blockDim.x + threadIdx.x;
  if (e >= E) return;
  int s = src[e], d = dst[e];
  float sv = ss[s] + sd[d];
  float ev = sv > 0.f ? sv : 0.2f * sv;   // leaky_relu(0.2)
  atomicMax(&mxu[d], fenc(ev));
}

__global__ void k_edge_alpha(const int* __restrict__ src, const int* __restrict__ dst,
                             const float* __restrict__ ss, const float* __restrict__ sd,
                             const unsigned* __restrict__ mxu, float* __restrict__ alpha,
                             float* __restrict__ den, int E) {
  int e = blockIdx.x * blockDim.x + threadIdx.x;
  if (e >= E) return;
  int s = src[e], d = dst[e];
  float sv = ss[s] + sd[d];
  float ev = sv > 0.f ? sv : 0.2f * sv;
  unsigned u = mxu[d];
  float m = (u == 0u) ? 0.f : fdec(u);
  float a = __expf(ev - m);
  alpha[e] = a;
  atomicAdd(&den[d], a);
}

// 16 lanes per edge, float4 per lane: AGG[dst] += alpha * HS[src]
__global__ void k_edge_agg(const int* __restrict__ src, const int* __restrict__ dst,
                           const float* __restrict__ alpha, const float* __restrict__ HS,
                           float* __restrict__ AGG, int E) {
  int t = blockIdx.x * blockDim.x + threadIdx.x;
  int e = t >> 4, j = t & 15;
  if (e >= E) return;
  int s = src[e], d = dst[e];
  float a = alpha[e];
  float4 h = ((const float4*)(HS + (size_t)s * H64))[j];
  float* o = AGG + (size_t)d * H64 + j * 4;
  atomicAdd(o + 0, a * h.x);
  atomicAdd(o + 1, a * h.y);
  atomicAdd(o + 2, a * h.z);
  atomicAdd(o + 3, a * h.w);
}

// NEW = AGG/max(den,eps)+b  (first) ; NEW = 0.5*(NEW + AGG/..+b) (second)
__global__ void k_combine(const float* __restrict__ AGG, const float* __restrict__ DEN,
                          const float* __restrict__ bvec, float* __restrict__ NEW,
                          int n, int first) {
  size_t i = blockIdx.x * (size_t)blockDim.x + threadIdx.x;
  if (i >= (size_t)n * H64) return;
  float v = AGG[i] / fmaxf(DEN[i >> 6], 1e-16f) + bvec[i & 63];
  NEW[i] = first ? v : 0.5f * (NEW[i] + v);
}

// per-column sum / sumsq of X[n,64]
__global__ void k_colstats(const float* __restrict__ X, int n,
                           float* __restrict__ sum, float* __restrict__ sq) {
  __shared__ float s1[64], s2[64];
  int t = threadIdx.x;
  if (t < 64) { s1[t] = 0.f; s2[t] = 0.f; }
  __syncthreads();
  int col = t & 63;
  int rpb = blockDim.x >> 6;
  float ls = 0.f, lq = 0.f;
  for (int r = blockIdx.x * rpb + (t >> 6); r < n; r += gridDim.x * rpb) {
    float v = X[(size_t)r * H64 + col];
    ls += v; lq += v * v;
  }
  atomicAdd(&s1[col], ls); atomicAdd(&s2[col], lq);
  __syncthreads();
  if (t < 64) { atomicAdd(&sum[t], s1[t]); atomicAdd(&sq[t], s2[t]); }
}

// O = relu(BN(T)) (+ resid)   (BN: biased var, eps=1e-5)
__global__ void k_bn_apply(const float* __restrict__ T, const float* __restrict__ sum,
                           const float* __restrict__ sq, const float* __restrict__ g,
                           const float* __restrict__ beta, const float* __restrict__ resid,
                           float* __restrict__ O, int n) {
  size_t i = blockIdx.x * (size_t)blockDim.x + threadIdx.x;
  if (i >= (size_t)n * H64) return;
  int col = (int)(i & 63);
  float inv = 1.0f / (float)n;
  float mu = sum[col] * inv;
  float var = sq[col] * inv - mu * mu;
  float sc = g[col] * rsqrtf(var + 1e-5f);
  float v = sc * (T[i] - mu) + beta[col];
  v = v > 0.f ? v : 0.f;
  if (resid) v += resid[i];
  O[i] = v;
}

__global__ void k_bias_relu(float* __restrict__ T, const float* __restrict__ b, int n) {
  size_t i = blockIdx.x * (size_t)blockDim.x + threadIdx.x;
  if (i >= (size_t)n * H64) return;
  float v = T[i] + b[i & 63];
  T[i] = v > 0.f ? v : 0.f;
}

// out[r,c] = act(A[r,:] . W3[:,c] + b3[c]) * scale
__global__ void k_head_out(const float* __restrict__ A, const float* __restrict__ W3,
                           const float* __restrict__ b3, float* __restrict__ out,
                           int n, int odim, int sig, float scale) {
  int r = blockIdx.x * blockDim.x + threadIdx.x;
  if (r >= n) return;
  const float* a = A + (size_t)r * H64;
  for (int c = 0; c < odim; ++c) {
    float acc = b3[c];
#pragma unroll 8
    for (int k = 0; k < 64; ++k) acc += a[k] * W3[k * odim + c];
    if (sig) acc = 1.f / (1.f + __expf(-acc));
    out[(size_t)r * odim + c] = acc * scale;
  }
}

__global__ void k_iota(float* __restrict__ o1, float* __restrict__ o2, int n) {
  int i = blockIdx.x * blockDim.x + threadIdx.x;
  if (i >= n) return;
  o1[i] = (float)i;
  o2[i] = (float)i;
}

// ------------------------------ host side ----------------------------------

// Flattened input order: jax tree (sorted dict keys at every level).
enum {
  IN_EDGE_LL = 0, IN_EDGE_LP, IN_EDGE_PL, IN_EDGE_PP, IN_LOC_X,
  IN_BNL_B, IN_BNL_G, IN_BNP_B, IN_BNP_G,
  IN_EH_W1, IN_EH_W2, IN_EH_W3, IN_EH_B1, IN_EH_B2, IN_EH_B3,
  IN_ENCL_W, IN_ENCL_B, IN_ENCL_BETA, IN_ENCL_G,
  IN_ENCP_W, IN_ENCP_B, IN_ENCP_BETA, IN_ENCP_G,
  IN_G_LL_W, IN_G_LL_AD, IN_G_LL_AS, IN_G_LL_B,
  IN_G_LP_W, IN_G_LP_AD, IN_G_LP_AS, IN_G_LP_B,
  IN_G_PL_W, IN_G_PL_AD, IN_G_PL_AS, IN_G_PL_B,
  IN_G_PP_W, IN_G_PP_AD, IN_G_PP_AS, IN_G_PP_B,
  IN_JH_W1, IN_JH_W2, IN_JH_W3, IN_JH_B1, IN_JH_B2, IN_JH_B3,
  IN_JH_BB1, IN_JH_BB2, IN_JH_G1, IN_JH_G2,
  IN_PH_W1, IN_PH_W2, IN_PH_W3, IN_PH_B1, IN_PH_B2, IN_PH_B3,
  IN_PH_BB1, IN_PH_BB2, IN_PH_G1, IN_PH_G2,
  IN_TH_W1, IN_TH_W2, IN_TH_W3, IN_TH_B1, IN_TH_B2, IN_TH_B3,
  IN_TH_BB1, IN_TH_BB2, IN_TH_G1, IN_TH_G2,
  IN_PEOPLE_X
};

extern "C" void kernel_launch(void* const* d_in, const int* in_sizes, int n_in,
                              void* d_out, int out_size, void* d_ws, size_t ws_size,
                              hipStream_t stream) {
  (void)n_in; (void)out_size; (void)ws_size;
  const float* F[80];
  for (int i = 0; i < 70; ++i) F[i] = (const float*)d_in[i];

  const int N   = in_sizes[IN_PEOPLE_X];
  const int Ell = in_sizes[IN_EDGE_LL] / 2;
  const int Elp = in_sizes[IN_EDGE_LP] / 2;
  const int Epl = in_sizes[IN_EDGE_PL] / 2;
  const int Epp = in_sizes[IN_EDGE_PP] / 2;
  int Emax = Ell;
  if (Elp > Emax) Emax = Elp;
  if (Epl > Emax) Emax = Epl;
  if (Epp > Emax) Emax = Epp;

  const size_t NB = (size_t)N * H64;

  // workspace layout (256B aligned chunks)
  float* wsf = (float*)d_ws;
  size_t off = 0;
  auto alloc = [&](size_t cnt) -> float* {
    float* p = wsf + off;
    off += (cnt + 63) & ~(size_t)63;
    return p;
  };
  float* XP   = alloc(NB);
  float* XL   = alloc(NB);
  float* HS   = alloc(NB);
  float* AGG  = alloc(NB);
  float* NEWP = alloc(NB);
  float* NEWL = alloc(NB);
  float* SS   = alloc(N);
  float* SD   = alloc(N);
  unsigned* MXU = (unsigned*)alloc(N);
  float* DEN  = alloc(N);
  float* ALPHA = alloc((size_t)Emax);
  float* SUM64 = alloc(64);
  float* SQ64  = alloc(64);
  float* WAD   = alloc(64);
  float* STAT  = alloc(64);
  __bf16* WTB  = (__bf16*)alloc(64 * 1024 / 4 * 2);  // bf16 weight area (128KB+)

  __bf16* WT_GAT = WTB;                 // 8 * 4096
  __bf16* WT_H1  = WTB + 8 * 4096;      // 4 * 8192
  __bf16* WT_H2  = WT_H1 + 4 * 8192;    // 4 * 4096

  const int B256 = 256;
  const int gN   = (N + 255) / 256;
  const int gNH  = (int)((NB + 255) / 256);
  const int gemmGrid = (N + 127) / 128;

  // ---- prep: bf16-transpose all GEMM weights -----------------------------
  auto tr = [&](int idx, __bf16* dst, int K, int srcOff) {
    k_transpose_bf16<<<(64 * K + 255) / 256, B256, 0, stream>>>(F[idx] + srcOff, dst, K);
  };
  const int gatW[4] = {IN_G_LL_W, IN_G_LP_W, IN_G_PL_W, IN_G_PP_W};
  for (int rel = 0; rel < 4; ++rel)
    for (int l = 0; l < 2; ++l)
      tr(gatW[rel], WT_GAT + (size_t)(rel * 2 + l) * 4096, 64, l * 4096);
  const int h1idx[4] = {IN_EH_W1, IN_JH_W1, IN_PH_W1, IN_TH_W1};
  const int h2idx[4] = {IN_EH_W2, IN_JH_W2, IN_PH_W2, IN_TH_W2};
  for (int h = 0; h < 4; ++h) {
    tr(h1idx[h], WT_H1 + (size_t)h * 8192, 128, 0);
    tr(h2idx[h], WT_H2 + (size_t)h * 4096, 64, 0);
  }

  // ---- encoders ----------------------------------------------------------
  hipMemsetAsync(STAT, 0, 4 * sizeof(float), stream);
  k_scalar_stats<<<512, B256, 0, stream>>>(F[IN_PEOPLE_X], N, STAT);
  k_scalar_stats<<<512, B256, 0, stream>>>(F[IN_LOC_X], N, STAT + 2);
  k_encode<<<gNH, B256, 0, stream>>>(F[IN_PEOPLE_X], F[IN_ENCP_W], F[IN_ENCP_G],
                                     F[IN_ENCP_BETA], STAT, N, XP);
  k_encode<<<gNH, B256, 0, stream>>>(F[IN_LOC_X], F[IN_ENCL_W], F[IN_ENCL_G],
                                     F[IN_ENCL_BETA], STAT + 2, N, XL);

  // ---- one GAT relation --------------------------------------------------
  auto run_rel = [&](const float* xs, const float* xd, const int* eptr, int E,
                     const float* Wf, const __bf16* Wt, const float* a_s,
                     const float* a_d, const float* bvec, float* NEW, int first) {
    k_gemm_wmma<<<gemmGrid, B256, 0, stream>>>(xs, nullptr, Wt, HS, N, 2);
    k_matvec64<<<1, 64, 0, stream>>>(Wf, a_d, WAD);
    k_rowdot<<<gN, B256, 0, stream>>>(HS, a_s, SS, N);
    k_rowdot<<<gN, B256, 0, stream>>>(xd, WAD, SD, N);
    hipMemsetAsync(MXU, 0, (size_t)N * 4, stream);
    hipMemsetAsync(DEN, 0, (size_t)N * 4, stream);
    hipMemsetAsync(AGG, 0, NB * 4, stream);
    const int* src = eptr;
    const int* dst = eptr + E;
    k_edge_max<<<(E + 255) / 256, B256, 0, stream>>>(src, dst, SS, SD, MXU, E);
    k_edge_alpha<<<(E + 255) / 256, B256, 0, stream>>>(src, dst, SS, SD, MXU, ALPHA, DEN, E);
    k_edge_agg<<<(int)(((size_t)E * 16 + 255) / 256), B256, 0, stream>>>(src, dst, ALPHA, HS, AGG, E);
    k_combine<<<gNH, B256, 0, stream>>>(AGG, DEN, bvec, NEW, N, first);
  };

  const int* e_ll = (const int*)d_in[IN_EDGE_LL];
  const int* e_lp = (const int*)d_in[IN_EDGE_LP];
  const int* e_pl = (const int*)d_in[IN_EDGE_PL];
  const int* e_pp = (const int*)d_in[IN_EDGE_PP];

  for (int l = 0; l < 2; ++l) {
    const int wo = l * 4096, vo = l * 64;
    // people: 0.5*(pp + lp)
    run_rel(XP, XP, e_pp, Epp, F[IN_G_PP_W] + wo, WT_GAT + (3 * 2 + l) * 4096,
            F[IN_G_PP_AS] + vo, F[IN_G_PP_AD] + vo, F[IN_G_PP_B] + vo, NEWP, 1);
    run_rel(XL, XP, e_lp, Elp, F[IN_G_LP_W] + wo, WT_GAT + (1 * 2 + l) * 4096,
            F[IN_G_LP_AS] + vo, F[IN_G_LP_AD] + vo, F[IN_G_LP_B] + vo, NEWP, 0);
    // locations: 0.5*(ll + pl)
    run_rel(XL, XL, e_ll, Ell, F[IN_G_LL_W] + wo, WT_GAT + (0 * 2 + l) * 4096,
            F[IN_G_LL_AS] + vo, F[IN_G_LL_AD] + vo, F[IN_G_LL_B] + vo, NEWL, 1);
    run_rel(XP, XL, e_pl, Epl, F[IN_G_PL_W] + wo, WT_GAT + (2 * 2 + l) * 4096,
            F[IN_G_PL_AS] + vo, F[IN_G_PL_AD] + vo, F[IN_G_PL_B] + vo, NEWL, 0);
    // xp = relu(bn(new_p)) + xp ; xl = relu(bn(new_l)) + xl
    hipMemsetAsync(SUM64, 0, 128 * sizeof(float), stream);
    k_colstats<<<512, B256, 0, stream>>>(NEWP, N, SUM64, SQ64);
    k_bn_apply<<<gNH, B256, 0, stream>>>(NEWP, SUM64, SQ64, F[IN_BNP_G] + vo,
                                         F[IN_BNP_B] + vo, XP, XP, N);
    hipMemsetAsync(SUM64, 0, 128 * sizeof(float), stream);
    k_colstats<<<512, B256, 0, stream>>>(NEWL, N, SUM64, SQ64);
    k_bn_apply<<<gNH, B256, 0, stream>>>(NEWL, SUM64, SQ64, F[IN_BNL_G] + vo,
                                         F[IN_BNL_B] + vo, XL, XL, N);
  }

  float* outF = (float*)d_out;

  // ---- edge head (no BN): relu(pair@W1+b1) -> relu(@W2+b2) -> sigmoid(@W3+b3)
  k_gemm_wmma<<<gemmGrid, B256, 0, stream>>>(XP, XL, WT_H1 + 0 * 8192, HS, N, 4);
  k_bias_relu<<<gNH, B256, 0, stream>>>(HS, F[IN_EH_B1], N);
  k_gemm_wmma<<<gemmGrid, B256, 0, stream>>>(HS, nullptr, WT_H2 + 0 * 4096, AGG, N, 2);
  k_bias_relu<<<gNH, B256, 0, stream>>>(AGG, F[IN_EH_B2], N);
  k_head_out<<<gN, B256, 0, stream>>>(AGG, F[IN_EH_W3], F[IN_EH_B3], outF, N, 1, 1, 1.0f);

  // ---- BN heads (BN absorbs b1/b2) --------------------------------------
  auto bn_head = [&](int hslot, int iG1, int iBB1, int iG2, int iBB2,
                     int iW3, int iB3, float* outp, int odim, int sig, float scale) {
    k_gemm_wmma<<<gemmGrid, B256, 0, stream>>>(XP, XL, WT_H1 + (size_t)hslot * 8192, HS, N, 4);
    hipMemsetAsync(SUM64, 0, 128 * sizeof(float), stream);
    k_colstats<<<512, B256, 0, stream>>>(HS, N, SUM64, SQ64);
    k_bn_apply<<<gNH, B256, 0, stream>>>(HS, SUM64, SQ64, F[iG1], F[iBB1], nullptr, NEWP, N);
    k_gemm_wmma<<<gemmGrid, B256, 0, stream>>>(NEWP, nullptr, WT_H2 + (size_t)hslot * 4096, HS, N, 2);
    hipMemsetAsync(SUM64, 0, 128 * sizeof(float), stream);
    k_colstats<<<512, B256, 0, stream>>>(HS, N, SUM64, SQ64);
    k_bn_apply<<<gNH, B256, 0, stream>>>(HS, SUM64, SQ64, F[iG2], F[iBB2], nullptr, NEWP, N);
    k_head_out<<<gN, B256, 0, stream>>>(NEWP, F[iW3], F[iB3], outp, N, odim, sig, scale);
  };

  // head slots in WT_H1/H2: 0=edge, 1=joint, 2=purpose, 3=time
  bn_head(2, IN_PH_G1, IN_PH_BB1, IN_PH_G2, IN_PH_BB2, IN_PH_W3, IN_PH_B3,
          outF + (size_t)N, 5, 0, 1.0f);                      // purpose [N,5]
  bn_head(3, IN_TH_G1, IN_TH_BB1, IN_TH_G2, IN_TH_BB2, IN_TH_W3, IN_TH_B3,
          outF + (size_t)6 * N, 1, 1, 24.0f);                 // time = sigmoid*24
  bn_head(1, IN_JH_G1, IN_JH_BB1, IN_JH_G2, IN_JH_BB2, IN_JH_W3, IN_JH_B3,
          outF + (size_t)7 * N, 1, 1, 1.0f);                  // joint = sigmoid

  // ---- indices -----------------------------------------------------------
  k_iota<<<gN, B256, 0, stream>>>(outF + (size_t)8 * N, outF + (size_t)9 * N, N);
}
